// HIEMLayer_21818433864128
// MI455X (gfx1250) — matrix operand order repaired and checked by reading, not compiled, not run on hardware
//
#include <hip/hip_runtime.h>

#define BATCH  512
#define LSEQ   256
#define DMODEL 512

typedef __attribute__((ext_vector_type(16))) __bf16 v16bf;
typedef __attribute__((ext_vector_type(8)))  float  v8f;

union AFrag { v16bf v; unsigned u[8]; };

__device__ inline unsigned short f2bf(float f) {
    union { float f; unsigned u; } c; c.f = f;
    unsigned u = c.u;
    return (unsigned short)((u + 0x7FFFu + ((u >> 16) & 1u)) >> 16);
}
// pack two floats into a bf16x2 dword (lo, hi)
__device__ inline unsigned f2bf2(float lo, float hi) {
    return (unsigned)f2bf(lo) | ((unsigned)f2bf(hi) << 16);
}
__device__ inline float bf2f(unsigned short h) {
    union { unsigned u; float f; } c; c.u = ((unsigned)h) << 16;
    return c.f;
}
// order-preserving float<->uint mapping so we can use ds_max_u32 atomics
__device__ inline unsigned f2ord(float f) {
    unsigned u = __float_as_uint(f);
    return (u & 0x80000000u) ? ~u : (u | 0x80000000u);
}
__device__ inline float ord2f(unsigned o) {
    unsigned u = (o & 0x80000000u) ? (o & 0x7FFFFFFFu) : ~o;
    return __uint_as_float(u);
}

// CDNA5 async copy: global -> LDS, 16 bytes, no VGPR round trip (ASYNCcnt).
__device__ inline void async_ld_b128(void* ldsPtr, unsigned gByteOff, const void* base) {
    unsigned ldsOff = (unsigned)(unsigned long long)ldsPtr;  // low 32 bits = LDS offset
    asm volatile("global_load_async_to_lds_b128 %0, %1, %2 offset:0"
                 :: "v"(ldsOff), "v"(gByteOff),
                    "s"((unsigned long long)base)
                 : "memory");
}
__device__ inline void wait_async0() {
    asm volatile("s_wait_asynccnt 0x0" ::: "memory");
}

// ---------------------------------------------------------------------------
// K1: P = X @ W   (X = concat(i, j) as [2*B*L, D] f32; P bf16)
// 256 threads = 8 waves, tile 128 rows x 128 cols, K stepped by 32.
// ---------------------------------------------------------------------------
__global__ __launch_bounds__(256) void k_proj(
    const float* __restrict__ X0, const float* __restrict__ X1,
    const float* __restrict__ W, unsigned short* __restrict__ P)
{
    __shared__ __attribute__((aligned(16))) unsigned short lA[128][32];      // [row][k]
    __shared__ __attribute__((aligned(16))) unsigned short lB[128][32 + 2];  // [n][k] (W^T slice)

    const int tid  = threadIdx.x;
    const int wave = tid >> 5, lane = tid & 31;
    const int m_lane = lane & 15, khalf = lane >> 4;
    const long rowBase = (long)blockIdx.x * 128;
    const int  colBase = blockIdx.y * 128;

    v8f acc[8] = {};

    for (int kk = 0; kk < DMODEL; kk += 32) {
        {   // stage A rows (f32 -> bf16x2 packed dword stores)
            int r  = tid >> 1;
            int ko = (tid & 1) * 16;
            long grow = rowBase + r;
            const float* src = (grow < (long)BATCH * LSEQ)
                ? (X0 + grow * DMODEL) : (X1 + (grow - (long)BATCH * LSEQ) * DMODEL);
            #pragma unroll
            for (int q = 0; q < 8; ++q)
                *(unsigned*)&lA[r][ko + 2 * q] =
                    f2bf2(src[kk + ko + 2 * q], src[kk + ko + 2 * q + 1]);
        }
        {   // stage B = W[kk..kk+32][colBase..+128] transposed into [n][k]
            int k  = tid >> 3;
            int no = (tid & 7) * 16;
            const float* src = W + (long)(kk + k) * DMODEL + colBase + no;
            #pragma unroll
            for (int q = 0; q < 16; ++q) lB[no + q][k] = f2bf(src[q]);
        }
        __syncthreads();

        AFrag a;
        #pragma unroll
        for (int p = 0; p < 8; ++p) {
            int kb = 2 * p + 16 * khalf;
            a.u[p] = *(const unsigned*)&lA[wave * 16 + m_lane][kb];
        }
        AFrag bfr[2];
        #pragma unroll
        for (int v = 0; v < 8; ++v)
            bfr[0].u[v] = *(const unsigned*)&lB[m_lane][2 * v + 16 * khalf];
        #pragma unroll
        for (int t = 0; t < 8; ++t) {
            if (t < 7) {
                int n = (t + 1) * 16 + m_lane;
                #pragma unroll
                for (int v = 0; v < 8; ++v)
                    bfr[(t + 1) & 1].u[v] = *(const unsigned*)&lB[n][2 * v + 16 * khalf];
            }
            acc[t] = __builtin_amdgcn_wmma_f32_16x16x32_bf16(
                false, a.v, false, bfr[t & 1].v, (short)0, acc[t], false, false);
        }
        __syncthreads();
    }

    #pragma unroll
    for (int t = 0; t < 8; ++t) {
        int n = colBase + t * 16 + m_lane;
        #pragma unroll
        for (int r = 0; r < 8; ++r) {
            long row = rowBase + wave * 16 + r + 8 * khalf;
            P[row * DMODEL + n] = f2bf(acc[t][r]);
        }
    }
}

// ---------------------------------------------------------------------------
// K2: S = P_x @ y^T  (256x256 per (side,batch)) fused with axis-1 (over l)
// softmax; writes normalized bf16 scores. 512 threads = 16 waves, each wave
// owns a 16-row strip. A-tiles staged via async global->LDS copy.
// ---------------------------------------------------------------------------
__global__ __launch_bounds__(512) void k_scores(
    const unsigned short* __restrict__ P,
    const float* __restrict__ I, const float* __restrict__ J,
    unsigned short* __restrict__ Sn)
{
    __shared__ __attribute__((aligned(16))) unsigned short lA[256][32];      // P rows [l][k]
    __shared__ __attribute__((aligned(16))) unsigned short lY[256][32 + 2];  // y rows [m][k]
    __shared__ unsigned colmaxU[256];
    __shared__ float    colsum[256];

    const int s = blockIdx.x >> 9;
    const int b = blockIdx.x & 511;
    const float* Y = (s == 0 ? J : I) + (long)b * LSEQ * DMODEL;
    const unsigned short* Px = P + ((long)s * BATCH + b) * LSEQ * DMODEL;

    const int tid  = threadIdx.x;
    const int wave = tid >> 5, lane = tid & 31;
    const int m_lane = lane & 15, khalf = lane >> 4;

    if (tid < 256) { colmaxU[tid] = 0u; colsum[tid] = 0.f; }

    v8f acc[16] = {};

    for (int kk = 0; kk < DMODEL; kk += 32) {
        {   // stage A: async bf16 copy of P rows (2x16B per thread)
            int r  = tid >> 1;
            int ko = (tid & 1) * 16;
            unsigned gb = (unsigned)(((long)r * DMODEL + kk + ko) * 2);
            async_ld_b128(&lA[r][ko],     gb,      Px);
            async_ld_b128(&lA[r][ko + 8], gb + 16, Px);
        }
        {   // stage Y rows (f32 -> bf16x2 packed), [m][k]
            int m  = tid >> 1;
            int ko = (tid & 1) * 16;
            const float* src = Y + (long)m * DMODEL + kk + ko;
            #pragma unroll
            for (int q = 0; q < 8; ++q)
                *(unsigned*)&lY[m][ko + 2 * q] = f2bf2(src[2 * q], src[2 * q + 1]);
        }
        wait_async0();
        __syncthreads();

        AFrag a;
        #pragma unroll
        for (int p = 0; p < 8; ++p) {
            int kb = 2 * p + 16 * khalf;
            a.u[p] = *(const unsigned*)&lA[wave * 16 + m_lane][kb];
        }
        AFrag bfr[2];
        #pragma unroll
        for (int v = 0; v < 8; ++v)
            bfr[0].u[v] = *(const unsigned*)&lY[m_lane][2 * v + 16 * khalf];
        #pragma unroll
        for (int t = 0; t < 16; ++t) {
            if (t < 15) {
                int m = (t + 1) * 16 + m_lane;
                #pragma unroll
                for (int v = 0; v < 8; ++v)
                    bfr[(t + 1) & 1].u[v] = *(const unsigned*)&lY[m][2 * v + 16 * khalf];
            }
            acc[t] = __builtin_amdgcn_wmma_f32_16x16x32_bf16(
                false, a.v, false, bfr[t & 1].v, (short)0, acc[t], false, false);
        }
        __syncthreads();
    }

    // ---- softmax over l (rows), per column m ----
    #pragma unroll
    for (int t = 0; t < 16; ++t) {
        float mx = acc[t][0];
        #pragma unroll
        for (int r = 1; r < 8; ++r) mx = fmaxf(mx, acc[t][r]);
        atomicMax(&colmaxU[t * 16 + m_lane], f2ord(mx));
    }
    __syncthreads();
    #pragma unroll
    for (int t = 0; t < 16; ++t) {
        int m = t * 16 + m_lane;
        float cm = ord2f(colmaxU[m]);
        float lsum = 0.f;
        #pragma unroll
        for (int r = 0; r < 8; ++r) {
            float e = __expf(acc[t][r] - cm);
            acc[t][r] = e;
            lsum += e;
        }
        atomicAdd(&colsum[m], lsum);
    }
    __syncthreads();
    #pragma unroll
    for (int t = 0; t < 16; ++t) {
        int m = t * 16 + m_lane;
        float inv = 1.f / colsum[m];
        #pragma unroll
        for (int r = 0; r < 8; ++r) {
            int l = wave * 16 + r + 8 * khalf;
            Sn[(((long)s * BATCH + b) * LSEQ + l) * LSEQ + m] = f2bf(acc[t][r] * inv);
        }
    }
}

// ---------------------------------------------------------------------------
// K3: a = Sn @ y   (256x512 per (side,batch), split into two N=256 halves)
// 512 threads = 16 waves x 16-row strips. A-tiles staged via async copy.
// ---------------------------------------------------------------------------
__global__ __launch_bounds__(512) void k_ctx(
    const unsigned short* __restrict__ Sn,
    const float* __restrict__ I, const float* __restrict__ J,
    unsigned short* __restrict__ Actx)
{
    __shared__ __attribute__((aligned(16))) unsigned short lS[256][32];      // Sn rows [l][m]
    __shared__ __attribute__((aligned(16))) unsigned short lY[256][32 + 2];  // [e][m] (y^T slice)

    const int bid  = blockIdx.x;
    const int half = bid & 1;
    const int b    = (bid >> 1) & 511;
    const int s    = bid >> 10;
    const float* Y = (s == 0 ? J : I) + (long)b * LSEQ * DMODEL;
    const unsigned short* A = Sn + ((long)s * BATCH + b) * LSEQ * LSEQ;
    unsigned short* Out = Actx + ((long)s * BATCH + b) * LSEQ * DMODEL + half * 256;

    const int tid  = threadIdx.x;
    const int wave = tid >> 5, lane = tid & 31;
    const int m_lane = lane & 15, khalf = lane >> 4;

    v8f acc[16] = {};

    for (int kk = 0; kk < LSEQ; kk += 32) {
        {   // stage A: async bf16 copy of Sn rows
            int r  = tid >> 1;
            int ko = (tid & 1) * 16;
            unsigned gb = (unsigned)(((long)r * LSEQ + kk + ko) * 2);
            async_ld_b128(&lS[r][ko],     gb,      A);
            async_ld_b128(&lS[r][ko + 8], gb + 16, A);
        }
        {   // stage B: lY[e][k] = y[kk+k][half*256+e]  (f32 -> bf16, transposed)
            int k  = tid >> 4;
            int eo = (tid & 15) * 16;
            const float* src = Y + (long)(kk + k) * DMODEL + half * 256 + eo;
            #pragma unroll
            for (int q = 0; q < 16; ++q) lY[eo + q][k] = f2bf(src[q]);
        }
        wait_async0();
        __syncthreads();

        AFrag a;
        #pragma unroll
        for (int p = 0; p < 8; ++p) {
            int kb = 2 * p + 16 * khalf;
            a.u[p] = *(const unsigned*)&lS[wave * 16 + m_lane][kb];
        }
        AFrag bfr[2];
        #pragma unroll
        for (int v = 0; v < 8; ++v)
            bfr[0].u[v] = *(const unsigned*)&lY[m_lane][2 * v + 16 * khalf];
        #pragma unroll
        for (int t = 0; t < 16; ++t) {
            if (t < 15) {
                int e = (t + 1) * 16 + m_lane;
                #pragma unroll
                for (int v = 0; v < 8; ++v)
                    bfr[(t + 1) & 1].u[v] = *(const unsigned*)&lY[e][2 * v + 16 * khalf];
            }
            acc[t] = __builtin_amdgcn_wmma_f32_16x16x32_bf16(
                false, a.v, false, bfr[t & 1].v, (short)0, acc[t], false, false);
        }
        __syncthreads();
    }

    #pragma unroll
    for (int t = 0; t < 16; ++t) {
        int e = t * 16 + m_lane;
        #pragma unroll
        for (int r = 0; r < 8; ++r) {
            int l = wave * 16 + r + 8 * khalf;
            Out[(long)l * DMODEL + e] = f2bf(acc[t][r]);
        }
    }
}

// ---------------------------------------------------------------------------
// K4: epilogue per (side,batch): out = [x-a, x*a]; beta = softmax(out . w);
// pooled = sum_l beta_l * out_l. 256 threads, coalesced column walks.
// ---------------------------------------------------------------------------
__global__ __launch_bounds__(256) void k_pool(
    const float* __restrict__ I, const float* __restrict__ J,
    const unsigned short* __restrict__ Actx,
    const float* __restrict__ wvec, float* __restrict__ outp)
{
    __shared__ float logit[256];
    __shared__ float red[8];
    __shared__ float beta[256];

    const int s = blockIdx.x >> 9;
    const int b = blockIdx.x & 511;
    const float* X = (s == 0 ? I : J) + (long)b * LSEQ * DMODEL;
    const unsigned short* A = Actx + ((long)s * BATCH + b) * LSEQ * DMODEL;

    const int t = threadIdx.x;
    logit[t] = 0.f;
    __syncthreads();

    const float w0 = wvec[t], w1 = wvec[t + 256], w2 = wvec[t + 512], w3 = wvec[t + 768];

    // phase A: beta logits
    for (int l = 0; l < LSEQ; ++l) {
        float x0 = X[(long)l * DMODEL + t],        x1 = X[(long)l * DMODEL + t + 256];
        float a0 = bf2f(A[(long)l * DMODEL + t]),  a1 = bf2f(A[(long)l * DMODEL + t + 256]);
        float p = (x0 - a0) * w0 + (x1 - a1) * w1 + (x0 * a0) * w2 + (x1 * a1) * w3;
        #pragma unroll
        for (int off = 16; off; off >>= 1) p += __shfl_xor(p, off, 32);
        if ((t & 31) == 0) atomicAdd(&logit[l], p);
    }
    __syncthreads();

    // softmax over l
    float mx = logit[t];
    #pragma unroll
    for (int off = 16; off; off >>= 1) mx = fmaxf(mx, __shfl_xor(mx, off, 32));
    if ((t & 31) == 0) red[t >> 5] = mx;
    __syncthreads();
    if (t == 0) {
        float m = red[0];
        for (int q = 1; q < 8; ++q) m = fmaxf(m, red[q]);
        red[0] = m;
    }
    __syncthreads();
    mx = red[0];
    float e = __expf(logit[t] - mx);
    float ssum = e;
    #pragma unroll
    for (int off = 16; off; off >>= 1) ssum += __shfl_xor(ssum, off, 32);
    __syncthreads();
    if ((t & 31) == 0) red[t >> 5] = ssum;
    __syncthreads();
    if (t == 0) {
        float s0 = 0.f;
        for (int q = 0; q < 8; ++q) s0 += red[q];
        red[0] = s0;
    }
    __syncthreads();
    beta[t] = e / red[0];
    __syncthreads();

    // phase B: pooled output
    float acc0 = 0.f, acc1 = 0.f, acc2 = 0.f, acc3 = 0.f;
    for (int l = 0; l < LSEQ; ++l) {
        float bl = beta[l];
        float x0 = X[(long)l * DMODEL + t],        x1 = X[(long)l * DMODEL + t + 256];
        float a0 = bf2f(A[(long)l * DMODEL + t]),  a1 = bf2f(A[(long)l * DMODEL + t + 256]);
        acc0 += bl * (x0 - a0); acc1 += bl * (x1 - a1);
        acc2 += bl * (x0 * a0); acc3 += bl * (x1 * a1);
    }
    float* o = outp + ((long)s * BATCH + b) * (2 * DMODEL);
    o[t] = acc0; o[t + 256] = acc1; o[t + 512] = acc2; o[t + 768] = acc3;
}

// ---------------------------------------------------------------------------
extern "C" void kernel_launch(void* const* d_in, const int* in_sizes, int n_in,
                              void* d_out, int out_size, void* d_ws, size_t ws_size,
                              hipStream_t stream) {
    const float* I = (const float*)d_in[0];
    const float* J = (const float*)d_in[1];
    const float* W = (const float*)d_in[2];
    const float* w = (const float*)d_in[3];
    float* out = (float*)d_out;

    // workspace layout (bf16): P [2*B*L*D] (reused as a after k_scores),
    // Sn [2*B*L*L]. Total = 256MB + 128MB = 384MB.
    unsigned short* P  = (unsigned short*)d_ws;
    unsigned short* Sn = P + (size_t)2 * BATCH * LSEQ * DMODEL;
    unsigned short* Ax = P;  // context overwrites P (dead after k_scores)

    k_proj<<<dim3((2 * BATCH * LSEQ) / 128, DMODEL / 128), 256, 0, stream>>>(I, J, W, P);
    k_scores<<<dim3(2 * BATCH), 512, 0, stream>>>(P, I, J, Sn);
    k_ctx<<<dim3(2 * BATCH * 2), 512, 0, stream>>>(Sn, I, J, Ax);
    k_pool<<<dim3(2 * BATCH), 256, 0, stream>>>(I, J, Ax, w, out);
}
